// HSMSSD_89867895701992
// MI455X (gfx1250) — compile-verified, matrix operand order checked by reading, not run on hardware
//
#include <hip/hip_runtime.h>
#include <cstdint>

// ---------------------------------------------------------------------------
// CDNA5 (gfx1250) bf16 WMMA types
// ---------------------------------------------------------------------------
typedef __attribute__((ext_vector_type(16))) __bf16 v16bf;
typedef __attribute__((ext_vector_type(8)))  float  v8f;

union Frag16 { uint32_t u[8]; uint4 q[2]; v16bf v; };
union AccF32 { float f[8]; v8f v; };

__device__ __forceinline__ uint16_t f2bf(float f) {
  union { __bf16 b; uint16_t u; } cv;
  cv.b = (__bf16)f;
  return cv.u;
}

// Fragment layout (16-bit operands, wave32):
//   lane = (half = lane>>4, lc = lane&15); lane's 8 dwords cover K =
//   {half*8 .. half*8+7} and {16+half*8 .. 16+half*8+7} of one row (A) /
//   one column (B). With tiles stored row-major [row][40] in LDS (A rows,
//   B^T rows = B columns), each fragment is two 16-byte contiguous reads.
__device__ __forceinline__ void load_frag(const uint16_t* s, int row, int half,
                                          Frag16& f) {
  const uint4* p = reinterpret_cast<const uint4*>(s + row * 40 + half * 8);
  f.q[0] = p[0];   // K = half*8 .. +7      (dwords 0..3)
  f.q[1] = p[2];   // K = 16+half*8 .. +7   (dwords 4..7)
}

// ---------------------------------------------------------------------------
// f32 -> bf16 conversion (grid-stride)
// ---------------------------------------------------------------------------
__global__ __launch_bounds__(256)
void cvt_f32_bf16_kernel(const float* __restrict__ src, uint16_t* __restrict__ dst,
                         long long n) {
  long long i = (long long)blockIdx.x * blockDim.x + threadIdx.x;
  const long long stride = (long long)gridDim.x * blockDim.x;
  for (; i < n; i += stride) dst[i] = f2bf(src[i]);
}

// ---------------------------------------------------------------------------
// f32 [R][Cc] -> bf16 transposed [Cc][R], 32x32 LDS tiles, batched via Z
// ---------------------------------------------------------------------------
__global__ __launch_bounds__(256)
void transpose_cvt_kernel(const float* __restrict__ src, uint16_t* __restrict__ dst,
                          int R, int Cc) {
  __shared__ float tile[32][33];
  const int r0 = blockIdx.y * 32;
  const int c0 = blockIdx.x * 32;
  const long long base = (long long)blockIdx.z * R * Cc;
  const int tr = threadIdx.x >> 5;   // 0..7
  const int tc = threadIdx.x & 31;
  #pragma unroll
  for (int i = 0; i < 4; ++i)
    tile[tr + i * 8][tc] = src[base + (long long)(r0 + tr + i * 8) * Cc + c0 + tc];
  __syncthreads();
  #pragma unroll
  for (int i = 0; i < 4; ++i)
    dst[base + (long long)(c0 + tr + i * 8) * R + r0 + tc] = f2bf(tile[tc][tr + i * 8]);
}

// ---------------------------------------------------------------------------
// bf16 GEMM, TN form:  out[M,N] = A[M,K] * B[K,N] (+bias[M])
//   A  : row-major [M][K] bf16  (lda, batched via strideA; 0 = shared)
//   Bt : B^T row-major [N][K] bf16 (ldb = K-stride, batched via strideB)
//   out: f32 (outF) and/or bf16 (outB), row-major [M][N], batched strideC
// 256 threads = 8 waves; wave tile 16 x (BN/WCOLS); per 32-deep K step each
// wave issues NFRAG v_wmma_f32_16x16x32_bf16 with one reused A fragment.
// Software-pipelined, K-loop unrolled by 2 with hard-coded ping-pong buffers;
// staging items are assigned structurally (NA = BM/64 A-chunks, NB = BN/64
// B-chunks per thread) so there is no divergent A/B classification, and the
// global pointers are simply advanced by 64 elements per iteration.
// ---------------------------------------------------------------------------
template<int BM, int BN, int WCOLS>
__global__ __launch_bounds__(256)
void gemm_bf16_tn_kernel(const uint16_t* __restrict__ Aptr, long long strideA, int lda,
                         const uint16_t* __restrict__ Btp, long long strideB, int ldb,
                         const float* __restrict__ bias,
                         float* __restrict__ outF, uint16_t* __restrict__ outB,
                         long long strideC, int M, int N, int K) {
  constexpr int WROWS = 8 / WCOLS;
  static_assert(BM == WROWS * 16, "BM must equal WROWS*16");
  constexpr int WN    = BN / WCOLS;
  constexpr int NFRAG = WN / 16;
  constexpr int NA    = BM / 64;           // A uint4 chunks per thread
  constexpr int NB    = BN / 64;           // B uint4 chunks per thread
  static_assert(NA + NB == 3 && BM % 64 == 0 && BN % 64 == 0, "tile config");
  constexpr int HALF = (BM + BN) * 40;     // elements per ping-pong buffer
  __shared__ uint16_t smem[2 * HALF];      // [buf][A: BM*40 | B: BN*40]

  __builtin_assume(threadIdx.x < 256);
  const int tid  = threadIdx.x;
  const int lane = tid & 31;
  const int wv   = tid >> 5;
  const int wr   = wv / WCOLS;
  const int wc   = wv % WCOLS;
  const int half = lane >> 4;
  const int lc   = lane & 15;

  const int m0 = blockIdx.y * BM;
  const int n0 = blockIdx.x * BN;
  const uint16_t* Ab = Aptr + (long long)blockIdx.z * strideA;
  const uint16_t* Bb = Btp + (long long)blockIdx.z * strideB;

  // structural staging assignment: thread covers row (tid>>2) (+64) with the
  // 16-byte chunk at element (tid&3)*8 of the 32-deep K slice.
  const int trow = tid >> 2;          // 0..63
  const int tq   = (tid & 3) * 8;     // 0,8,16,24

  const uint16_t* gA0 = Ab + (long long)(m0 + trow) * lda + tq;
  const uint16_t* gA1 = gA0 + 64ll * lda;
  const uint16_t* gB0 = Bb + (long long)(n0 + trow) * ldb + tq;
  const uint16_t* gB1 = gB0 + 64ll * ldb;

  const int soA0 = trow * 40 + tq;
  const int soA1 = soA0 + 64 * 40;
  const int soB0 = BM * 40 + trow * 40 + tq;
  const int soB1 = soB0 + 64 * 40;

  AccF32 acc[NFRAG];
  #pragma unroll
  for (int j = 0; j < NFRAG; ++j)
    #pragma unroll
    for (int e = 0; e < 8; ++e) acc[j].f[e] = 0.0f;

  auto compute = [&](int bufbase) {
    Frag16 af;
    load_frag(&smem[bufbase], wr * 16 + lc, half, af);
    #pragma unroll
    for (int j = 0; j < NFRAG; ++j) {
      Frag16 bf;
      load_frag(&smem[bufbase + BM * 40], wc * WN + j * 16 + lc, half, bf);
      acc[j].v = __builtin_amdgcn_wmma_f32_16x16x32_bf16(
          false, af.v, false, bf.v, (short)0, acc[j].v, false, false);
    }
  };

  // prologue: fetch tile 0
  uint4 va0 = {}, va1 = {}, vb0 = {}, vb1 = {};
  va0 = *reinterpret_cast<const uint4*>(gA0);
  if constexpr (NA == 2) va1 = *reinterpret_cast<const uint4*>(gA1);
  vb0 = *reinterpret_cast<const uint4*>(gB0);
  if constexpr (NB == 2) vb1 = *reinterpret_cast<const uint4*>(gB1);

  for (int kk = 0; kk < K; kk += 64) {
    // ---------- K step kk : buffer 0 ----------
    *reinterpret_cast<uint4*>(&smem[soA0]) = va0;
    if constexpr (NA == 2) *reinterpret_cast<uint4*>(&smem[soA1]) = va1;
    *reinterpret_cast<uint4*>(&smem[soB0]) = vb0;
    if constexpr (NB == 2) *reinterpret_cast<uint4*>(&smem[soB1]) = vb1;
    __syncthreads();
    // mid-step tile at immediate offset +32 elements (always < K)
    va0 = *reinterpret_cast<const uint4*>(gA0 + 32);
    if constexpr (NA == 2) va1 = *reinterpret_cast<const uint4*>(gA1 + 32);
    vb0 = *reinterpret_cast<const uint4*>(gB0 + 32);
    if constexpr (NB == 2) vb1 = *reinterpret_cast<const uint4*>(gB1 + 32);
    __builtin_prefetch(gA0 + 96, 0, 3);   // speculative: safe if OOB
    __builtin_prefetch(gB0 + 96, 0, 3);
    compute(0);

    // ---------- K step kk+32 : buffer 1 ----------
    *reinterpret_cast<uint4*>(&smem[HALF + soA0]) = va0;
    if constexpr (NA == 2) *reinterpret_cast<uint4*>(&smem[HALF + soA1]) = va1;
    *reinterpret_cast<uint4*>(&smem[HALF + soB0]) = vb0;
    if constexpr (NB == 2) *reinterpret_cast<uint4*>(&smem[HALF + soB1]) = vb1;
    __syncthreads();
    if (kk + 64 < K) {
      gA0 += 64; gA1 += 64; gB0 += 64; gB1 += 64;
      va0 = *reinterpret_cast<const uint4*>(gA0);
      if constexpr (NA == 2) va1 = *reinterpret_cast<const uint4*>(gA1);
      vb0 = *reinterpret_cast<const uint4*>(gB0);
      if constexpr (NB == 2) vb1 = *reinterpret_cast<const uint4*>(gB1);
    }
    compute(HALF);
    // no trailing barrier: the next buffer-0 store happens after the buffer-1
    // barrier, which every wave reaches only after computing buffer 0.
  }

  // ---- epilogue: D VGPR i -> M = half*8 + i, N = lc ----
  #pragma unroll
  for (int j = 0; j < NFRAG; ++j) {
    const int col = n0 + wc * WN + j * 16 + lc;
    #pragma unroll
    for (int i = 0; i < 8; ++i) {
      const int row = m0 + wr * 16 + half * 8 + i;
      float v = acc[j].f[i];
      if (bias) v += bias[row];
      const long long idx = (long long)blockIdx.z * strideC + (long long)row * N + col;
      if (outF) outF[idx] = v;
      if (outB) outB[idx] = f2bf(v);
    }
  }
}

// ---------------------------------------------------------------------------
// depthwise 3x3 conv (pad=1) on [B*192, 64, 64]; one block per (b,ch) plane
// ---------------------------------------------------------------------------
__global__ __launch_bounds__(256)
void dwconv3x3_kernel(const float* __restrict__ in, const float* __restrict__ w,
                      const float* __restrict__ bias, float* __restrict__ out) {
  const int bc = blockIdx.x;           // b*192 + ch
  const int ch = bc % 192;
  float wk[9];
  #pragma unroll
  for (int i = 0; i < 9; ++i) wk[i] = w[ch * 9 + i];
  const float bb = bias[ch];
  const float* ip = in + (long long)bc * 4096;
  float* op = out + (long long)bc * 4096;
  for (int p = threadIdx.x; p < 4096; p += 256) {
    const int y = p >> 6, x = p & 63;
    float acc = bb;
    for (int dy = -1; dy <= 1; ++dy) {
      const int yy = y + dy;
      if (yy < 0 || yy > 63) continue;
      for (int dx = -1; dx <= 1; ++dx) {
        const int xx = x + dx;
        if (xx < 0 || xx > 63) continue;
        acc += wk[(dy + 1) * 3 + (dx + 1)] * ip[yy * 64 + xx];
      }
    }
    op[p] = acc;
  }
}

// ---------------------------------------------------------------------------
// softmax over L per (b,s) row of dt (+A[s]); AB = sm*Bm row-major [B][S][L]
// (A-operand of the h^T GEMM); Cm emitted TRANSPOSED [B][L][S] (B^T operand
// of the y GEMM). bcdt channels: {Bm:0-63, Cm:64-127, dt:128-191}
// ---------------------------------------------------------------------------
__global__ __launch_bounds__(256)
void softmax_ab_kernel(const float* __restrict__ bcdt, const float* __restrict__ Avec,
                       uint16_t* __restrict__ AB, uint16_t* __restrict__ CmT) {
  const int b = blockIdx.x >> 6;
  const int s = blockIdx.x & 63;
  const long long base = (long long)b * 192 * 4096;
  const float* Bm = bcdt + base + (long long)s * 4096;
  const float* Cm = bcdt + base + (long long)(64 + s) * 4096;
  const float* dt = bcdt + base + (long long)(128 + s) * 4096;
  const float a = Avec[s];
  const int tid = threadIdx.x;
  float vals[16];
  float mx = -3.4e38f;
  #pragma unroll
  for (int i = 0; i < 16; ++i) {
    vals[i] = dt[tid + i * 256] + a;
    mx = fmaxf(mx, vals[i]);
  }
  __shared__ float red[256];
  red[tid] = mx;
  __syncthreads();
  for (int off = 128; off > 0; off >>= 1) {
    if (tid < off) red[tid] = fmaxf(red[tid], red[tid + off]);
    __syncthreads();
  }
  mx = red[0];
  __syncthreads();
  float sum = 0.0f;
  #pragma unroll
  for (int i = 0; i < 16; ++i) { vals[i] = __expf(vals[i] - mx); sum += vals[i]; }
  red[tid] = sum;
  __syncthreads();
  for (int off = 128; off > 0; off >>= 1) {
    if (tid < off) red[tid] += red[tid + off];
    __syncthreads();
  }
  const float inv = 1.0f / red[0];
  const long long ob = ((long long)b * 64 + s) * 4096;
  const long long cb = (long long)b * 4096 * 64 + s;
  #pragma unroll
  for (int i = 0; i < 16; ++i) {
    const int l = tid + i * 256;
    AB[ob + l] = f2bf(vals[i] * inv * Bm[l]);
    CmT[cb + (long long)l * 64] = f2bf(Cm[l]);
  }
}

// ---------------------------------------------------------------------------
// hg^T[b][s][c] = h1 * (silu(z) + Dskip);  hz layout [B][1024][64]
// ---------------------------------------------------------------------------
__global__ __launch_bounds__(256)
void gate_kernel(const float* __restrict__ hz, const float* __restrict__ Dskip,
                 uint16_t* __restrict__ hgT) {
  const long long i = (long long)blockIdx.x * 256 + threadIdx.x;  // < B*64*512
  const long long b = i >> 15;
  const int s = (int)((i >> 9) & 63);
  const int c = (int)(i & 511);
  const float h1 = hz[(b << 16) + (long long)c * 64 + s];
  const float z  = hz[(b << 16) + 32768 + (long long)c * 64 + s];
  const float sig = 1.0f / (1.0f + __expf(-z));
  hgT[i] = f2bf(h1 * (z * sig + Dskip[0]));
}

// ---------------------------------------------------------------------------
// host-side orchestration
// ---------------------------------------------------------------------------
extern "C" void kernel_launch(void* const* d_in, const int* in_sizes, int n_in,
                              void* d_out, int out_size, void* d_ws, size_t ws_size,
                              hipStream_t stream) {
  (void)in_sizes; (void)n_in; (void)out_size; (void)ws_size;
  constexpr int  Bn = 32;
  constexpr int  C  = 512;
  constexpr int  L  = 4096;
  constexpr int  S  = 64;
  constexpr long long CL = (long long)C * L;

  const float* x      = (const float*)d_in[0];
  const float* w_bcdt = (const float*)d_in[1];
  const float* b_bcdt = (const float*)d_in[2];
  const float* w_dw   = (const float*)d_in[3];
  const float* b_dw   = (const float*)d_in[4];
  const float* w_hz   = (const float*)d_in[5];
  const float* b_hz   = (const float*)d_in[6];
  const float* w_out  = (const float*)d_in[7];
  const float* b_out  = (const float*)d_in[8];
  const float* Avec   = (const float*)d_in[9];
  const float* Dskip  = (const float*)d_in[10];

  char* ws = (char*)d_ws;
  size_t off = 0;
  auto alloc = [&](size_t bytes) {
    char* p = ws + off;
    off += (bytes + 255) & ~(size_t)255;
    return p;
  };
  uint16_t* xb      = (uint16_t*)alloc((size_t)Bn * CL * 2);   // x  [B][C][L] bf16
  uint16_t* xbT     = (uint16_t*)alloc((size_t)Bn * CL * 2);   // x^T [B][L][C] bf16
  uint16_t* wbcdt_b = (uint16_t*)alloc((size_t)192 * 512 * 2);
  uint16_t* whz_b   = (uint16_t*)alloc((size_t)1024 * 512 * 2);
  uint16_t* wout_b  = (uint16_t*)alloc((size_t)512 * 512 * 2);
  float*    bcdt0   = (float*)alloc((size_t)Bn * 192 * L * 4); // pre-conv
  float*    bcdt1   = (float*)alloc((size_t)Bn * 192 * L * 4); // post-conv
  uint16_t* ABb     = (uint16_t*)alloc((size_t)Bn * S * L * 2);  // [B][S][L]
  uint16_t* CmT     = (uint16_t*)alloc((size_t)Bn * L * S * 2);  // [B][L][S]
  uint16_t* hTb     = (uint16_t*)alloc((size_t)Bn * S * C * 2);  // [B][S][C]
  float*    hzf     = (float*)alloc((size_t)Bn * 1024 * 64 * 4); // [B][1024][64]
  uint16_t* hgTb    = (uint16_t*)alloc((size_t)Bn * S * C * 2);  // [B][S][C]
  uint16_t* hob     = (uint16_t*)alloc((size_t)Bn * C * S * 2);  // [B][C][S]

  float* y_out  = (float*)d_out;                  // [B,512,64,64]
  float* ho_out = y_out + (long long)Bn * C * L;  // [B,512,64]

  // 1. one-time bf16 conversions (+ x transpose for the bcdt GEMM's B^T)
  cvt_f32_bf16_kernel<<<8192, 256, 0, stream>>>(x, xb, (long long)Bn * CL);
  transpose_cvt_kernel<<<dim3(L / 32, C / 32, Bn), 256, 0, stream>>>(x, xbT, C, L);
  cvt_f32_bf16_kernel<<<96,  256, 0, stream>>>(w_bcdt, wbcdt_b, 192LL * 512);
  cvt_f32_bf16_kernel<<<512, 256, 0, stream>>>(w_hz,   whz_b,   1024LL * 512);
  cvt_f32_bf16_kernel<<<256, 256, 0, stream>>>(w_out,  wout_b,  512LL * 512);

  // 2. bcdt = W_bcdt @ x + b_bcdt   (M=192,N=4096,K=512; B^T = x^T)
  gemm_bf16_tn_kernel<64, 128, 2><<<dim3(L / 128, 192 / 64, Bn), 256, 0, stream>>>(
      wbcdt_b, 0, 512, xbT, CL, 512, b_bcdt, bcdt0, nullptr, 192LL * L, 192, L, 512);

  // 3. depthwise 3x3 + b_dw
  dwconv3x3_kernel<<<Bn * 192, 256, 0, stream>>>(bcdt0, w_dw, b_dw, bcdt1);

  // 4. softmax(dt + A) * Bm -> AB [B][S][L]; Cm -> Cm^T [B][L][S]
  softmax_ab_kernel<<<Bn * S, 256, 0, stream>>>(bcdt1, Avec, ABb, CmT);

  // 5. h^T = AB @ x^T  ->  [B][S][C]   (M=64,N=512,K=4096; A = AB, B^T = x)
  gemm_bf16_tn_kernel<64, 128, 2><<<dim3(C / 128, 1, Bn), 256, 0, stream>>>(
      ABb, (long long)S * L, L, xb, CL, L, nullptr, nullptr, hTb,
      (long long)S * C, S, C, L);

  // 6. hz = W_hz @ h + b_hz  [B][1024][64]  (M=1024,N=64,K=512; B^T = h^T)
  gemm_bf16_tn_kernel<128, 64, 1><<<dim3(1, 1024 / 128, Bn), 256, 0, stream>>>(
      whz_b, 0, 512, hTb, (long long)S * C, 512, b_hz, hzf, nullptr,
      1024LL * 64, 1024, 64, 512);

  // 7. hg^T = (h1 * (silu(z) + Dskip))^T  [B][S][C]
  gate_kernel<<<(Bn * S * C) / 256, 256, 0, stream>>>(hzf, Dskip, hgTb);

  // 8. ho = W_out @ hg + b_out -> f32 out + bf16 [B][C][S] (A of y-GEMM)
  gemm_bf16_tn_kernel<128, 64, 1><<<dim3(1, 512 / 128, Bn), 256, 0, stream>>>(
      wout_b, 0, 512, hgTb, (long long)S * C, 512, b_out, ho_out, hob,
      (long long)C * S, 512, 64, 512);

  // 9. y = ho @ Cm   [B][512][4096]  (M=512,N=4096,K=64; A = ho, B^T = Cm^T)
  gemm_bf16_tn_kernel<64, 128, 2><<<dim3(L / 128, 512 / 64, Bn), 256, 0, stream>>>(
      hob, (long long)C * S, 64, CmT, (long long)L * S, 64, nullptr, y_out, nullptr,
      CL, 512, L, 64);
}
